// MuModel_53506702573826
// MI455X (gfx1250) — matrix operand order repaired and checked by reading, not compiled, not run on hardware
//
#include <hip/hip_runtime.h>
#include <hip/hip_bf16.h>
#include <math.h>

#define B_SEQ   4
#define L_SEQ   2048
#define ROWS    (B_SEQ * L_SEQ)          // 8192
#define DMODEL  512
#define DINNER  1024
#define DXZ     (2 * DINNER)             // 2048
#define DTRANK  32
#define DSTATE  16
#define NDBL    (DTRANK + 2 * DSTATE)    // 64
#define NOUT    144

typedef __attribute__((ext_vector_type(16))) __bf16 bf16x16;
typedef __attribute__((ext_vector_type(8)))  float  v8f;

union FragU { uint4 q[2]; bf16x16 v; };

__device__ __forceinline__ unsigned short f32_to_bf16(float f) {
  unsigned int u = __builtin_bit_cast(unsigned int, f);
  u += 0x7FFFu + ((u >> 16) & 1u);          // round-to-nearest-even
  return (unsigned short)(u >> 16);
}

__device__ __forceinline__ float sigmoidf_(float x) {
  return 1.0f / (1.0f + __expf(-x));
}

// ---------------------------------------------------------------------------
// Generic bf16 WMMA GEMM: C[M,N] (f32) = A[M,K] (bf16) * Bt[N,K]^T (bf16).
// Each wave32 computes a 16 x (NT*16) tile with v_wmma_f32_16x16x32_bf16.
// A scheduling barrier between the load phase and the WMMA chain pins all
// fragment loads of a K-step before the first WMMA: b0..b3 live ranges then
// overlap, forcing distinct VGPRs, so the waitcnt pass emits PARTIAL
// s_wait_loadcnt values and WMMA t overlaps the in-flight loads of t+1..
// ---------------------------------------------------------------------------
template<int NT>
__global__ __launch_bounds__(256) void wmma_gemm_bf16_k(
    const unsigned short* __restrict__ A, const unsigned short* __restrict__ Bt,
    float* __restrict__ C, int Mtiles, int NtGroups, int K,
    int lda, int ldb, int ldc) {
  const int lane = threadIdx.x & 31;
  const int wave = blockIdx.x * (blockDim.x >> 5) + (threadIdx.x >> 5);
  if (wave >= Mtiles * NtGroups) return;           // wave-uniform guard
  const int tm = wave / NtGroups;
  const int tg = wave - tm * NtGroups;
  const int m0 = tm << 4;
  const int n0 = tg * (NT << 4);

  const int half = lane >> 4;
  const int sub  = lane & 15;

  const unsigned short* aPtr = A  + (size_t)(m0 + sub) * lda + (half << 4);
  const unsigned short* bPtr = Bt + (size_t)(n0 + sub) * ldb + (half << 4);

  const v8f zero = {0.f, 0.f, 0.f, 0.f, 0.f, 0.f, 0.f, 0.f};
  v8f acc[NT];
#pragma unroll
  for (int t = 0; t < NT; ++t) acc[t] = zero;

  for (int k0 = 0; k0 < K; k0 += 32) {
    // ---- load phase: A fragment + NT B fragments, distinct registers ----
    FragU a, b[NT];
    a.q[0] = *(const uint4*)(aPtr + k0);
    a.q[1] = *(const uint4*)(aPtr + k0 + 8);
#pragma unroll
    for (int t = 0; t < NT; ++t) {
      const unsigned short* bp = bPtr + (size_t)t * ((size_t)ldb << 4) + k0;
      b[t].q[0] = *(const uint4*)(bp);
      b[t].q[1] = *(const uint4*)(bp + 8);
    }
    __builtin_prefetch(aPtr + k0 + 32, 0, 3);      // global_prefetch_b8 (SE)
    // scheduling barrier: keep all loads above the WMMA chain
    asm volatile("" ::: "memory");
    // ---- math phase: WMMA chain; partial load waits overlap the tail ----
#pragma unroll
    for (int t = 0; t < NT; ++t) {
      acc[t] = __builtin_amdgcn_wmma_f32_16x16x32_bf16(
          false, a.v, false, b[t].v, (short)0, acc[t], false, false);
    }
  }

#pragma unroll
  for (int t = 0; t < NT; ++t) {
    float* cPtr = C + (size_t)(m0 + (half << 3)) * ldc + n0 + (t << 4) + sub;
#pragma unroll
    for (int j = 0; j < 8; ++j) cPtr[(size_t)j * ldc] = acc[t][j];
  }
}

// W[K,N] f32 -> Wt[N,K] bf16 (one-time per launch, ~3 MB total)
__global__ __launch_bounds__(256) void transpose_bf16_kernel(
    const float* __restrict__ W, unsigned short* __restrict__ Wt, int K, int N) {
  int idx = blockIdx.x * blockDim.x + threadIdx.x;
  if (idx >= K * N) return;
  int n = idx / K;
  int k = idx - n * K;
  Wt[idx] = f32_to_bf16(W[(size_t)k * N + n]);
}

// h = tanh(x @ w0 + b0), stored bf16 for the in_proj WMMA GEMM
__global__ __launch_bounds__(256) void embed_kernel(
    const float* __restrict__ x, const float* __restrict__ w0,
    const float* __restrict__ b0, unsigned short* __restrict__ h_bf) {
  int idx = blockIdx.x * blockDim.x + threadIdx.x;
  if (idx >= ROWS * DMODEL) return;
  int r = idx >> 9;
  int c = idx & (DMODEL - 1);
  float v = b0[c];
  v += x[r * 3 + 0] * w0[0 * DMODEL + c];
  v += x[r * 3 + 1] * w0[1 * DMODEL + c];
  v += x[r * 3 + 2] * w0[2 * DMODEL + c];
  h_bf[idx] = f32_to_bf16(tanhf(v));
}

// causal depthwise conv (D_CONV=4) + SiLU; writes f32 (for scan) + bf16 (for GEMM)
__global__ __launch_bounds__(256) void conv_silu_kernel(
    const float* __restrict__ xz, const float* __restrict__ conv_w,
    const float* __restrict__ conv_b, float* __restrict__ xc,
    unsigned short* __restrict__ xc_bf) {
  int idx = blockIdx.x * blockDim.x + threadIdx.x;
  if (idx >= ROWS * DINNER) return;
  int r = idx >> 10;
  int d = idx & (DINNER - 1);
  int l = r & (L_SEQ - 1);
  float acc = conv_b[d];
#pragma unroll
  for (int k = 0; k < 4; ++k) {
    int l2 = l + k - 3;
    if (l2 >= 0) acc += xz[(size_t)(r + k - 3) * DXZ + d] * conv_w[d * 4 + k];
  }
  float s = acc * sigmoidf_(acc);
  xc[idx] = s;
  xc_bf[idx] = f32_to_bf16(s);
}

// dt slice of x_dbl -> bf16 A-matrix for dt_proj GEMM
__global__ __launch_bounds__(256) void slice_dt_bf16_kernel(
    const float* __restrict__ xdbl, unsigned short* __restrict__ dtin) {
  int idx = blockIdx.x * blockDim.x + threadIdx.x;
  if (idx >= ROWS * DTRANK) return;
  int r = idx >> 5;
  int j = idx & (DTRANK - 1);
  dtin[idx] = f32_to_bf16(xdbl[(size_t)r * NDBL + j]);
}

// in-place dt = softplus(dtp + dt_proj_b)
__global__ __launch_bounds__(256) void softplus_kernel(
    float* __restrict__ dtp, const float* __restrict__ dtb) {
  int idx = blockIdx.x * blockDim.x + threadIdx.x;
  if (idx >= ROWS * DINNER) return;
  int d = idx & (DINNER - 1);
  float v = dtp[idx] + dtb[d];
  dtp[idx] = (v > 20.0f) ? v : log1pf(__expf(v));
}

// Selective scan: lane = (b,d,n); fp32 state in registers; 16-lane shfl_xor
// reduction over states; fused +D*xc, SiLU(z) gate, bf16 output for out_proj.
__global__ __launch_bounds__(256) void scan_kernel(
    const float* __restrict__ dt, const float* __restrict__ xc,
    const float* __restrict__ xdbl, const float* __restrict__ xz,
    const float* __restrict__ A_log, const float* __restrict__ Dv,
    unsigned short* __restrict__ y_bf) {
  const int t = blockIdx.x * blockDim.x + threadIdx.x;
  const int n  = t & (DSTATE - 1);
  const int bd = t >> 4;
  if (bd >= B_SEQ * DINNER) return;
  const int d = bd & (DINNER - 1);
  const int b = bd >> 10;

  const float Acoef = -__expf(A_log[d * DSTATE + n]);
  const float Dd = Dv[d];
  float h = 0.0f;

  const float* pdt = dt   + (size_t)b * L_SEQ * DINNER + d;
  const float* pxc = xc   + (size_t)b * L_SEQ * DINNER + d;
  const float* pB  = xdbl + (size_t)b * L_SEQ * NDBL + DTRANK + n;
  const float* pC  = pB + DSTATE;
  const float* pz  = xz   + (size_t)b * L_SEQ * DXZ + DINNER + d;
  unsigned short* py = y_bf + (size_t)b * L_SEQ * DINNER + d;

  for (int l = 0; l < L_SEQ; ++l) {
    float dtv = *pdt;
    float xcv = *pxc;
    float Bv  = *pB;
    float Cv  = *pC;
    float dA  = __expf(dtv * Acoef);
    h = dA * h + dtv * Bv * xcv;
    float yp = h * Cv;
    yp += __shfl_xor(yp, 1, 32);
    yp += __shfl_xor(yp, 2, 32);
    yp += __shfl_xor(yp, 4, 32);
    yp += __shfl_xor(yp, 8, 32);
    if (n == 0) {
      float zv = *pz;
      float yv = (yp + Dd * xcv) * (zv * sigmoidf_(zv));
      *py = f32_to_bf16(yv);
    }
    pdt += DINNER; pxc += DINNER; pB += NDBL; pC += NDBL; pz += DXZ; py += DINNER;
  }
}

__global__ __launch_bounds__(256) void cvt_bf16_kernel(
    const float* __restrict__ in, unsigned short* __restrict__ out, int n) {
  int idx = blockIdx.x * blockDim.x + threadIdx.x;
  if (idx < n) out[idx] = f32_to_bf16(in[idx]);
}

// in-place softmax over rows of 144 (+b1 folded in); one wave32 per row
__global__ __launch_bounds__(256) void softmax_kernel(
    float* __restrict__ out, const float* __restrict__ b1) {
  const int wave = blockIdx.x * (blockDim.x >> 5) + (threadIdx.x >> 5);
  const int lane = threadIdx.x & 31;
  if (wave >= ROWS) return;
  float* row = out + (size_t)wave * NOUT;
  float vals[5];
  float mx = -3.4e38f;
#pragma unroll
  for (int i = 0; i < 5; ++i) {
    int c = lane + i * 32;
    float v = (c < NOUT) ? row[c] + b1[c] : -3.4e38f;
    vals[i] = v;
    mx = fmaxf(mx, v);
  }
  for (int m = 16; m >= 1; m >>= 1) mx = fmaxf(mx, __shfl_xor(mx, m, 32));
  float sum = 0.0f;
#pragma unroll
  for (int i = 0; i < 5; ++i) {
    int c = lane + i * 32;
    if (c < NOUT) { vals[i] = __expf(vals[i] - mx); sum += vals[i]; }
  }
  for (int m = 16; m >= 1; m >>= 1) sum += __shfl_xor(sum, m, 32);
  float inv = 1.0f / sum;
#pragma unroll
  for (int i = 0; i < 5; ++i) {
    int c = lane + i * 32;
    if (c < NOUT) row[c] = vals[i] * inv;
  }
}

#define CEILDIV(a, b) (((a) + (b) - 1) / (b))

extern "C" void kernel_launch(void* const* d_in, const int* in_sizes, int n_in,
                              void* d_out, int out_size, void* d_ws, size_t ws_size,
                              hipStream_t stream) {
  (void)in_sizes; (void)n_in; (void)out_size; (void)ws_size;
  const float* x        = (const float*)d_in[0];
  const float* w0       = (const float*)d_in[1];
  const float* b0       = (const float*)d_in[2];
  const float* in_proj  = (const float*)d_in[3];
  const float* conv_w   = (const float*)d_in[4];
  const float* conv_b   = (const float*)d_in[5];
  const float* x_proj   = (const float*)d_in[6];
  const float* dt_proj  = (const float*)d_in[7];
  const float* dt_pb    = (const float*)d_in[8];
  const float* A_log    = (const float*)d_in[9];
  const float* Dv       = (const float*)d_in[10];
  const float* out_proj = (const float*)d_in[11];
  const float* w1       = (const float*)d_in[12];
  const float* b1       = (const float*)d_in[13];
  float* out = (float*)d_out;

  char* ws = (char*)d_ws;
  size_t off = 0;
  auto alloc = [&](size_t bytes) -> char* {
    size_t o = (off + 255) & ~(size_t)255;
    off = o + bytes;
    return ws + o;
  };

  unsigned short* Wt_in  = (unsigned short*)alloc((size_t)DXZ * DMODEL * 2);     // 2048x512
  unsigned short* Wt_xp  = (unsigned short*)alloc((size_t)NDBL * DINNER * 2);    // 64x1024
  unsigned short* Wt_dt  = (unsigned short*)alloc((size_t)DINNER * DTRANK * 2);  // 1024x32
  unsigned short* Wt_op  = (unsigned short*)alloc((size_t)DMODEL * DINNER * 2);  // 512x1024
  unsigned short* Wt_w1  = (unsigned short*)alloc((size_t)NOUT * DMODEL * 2);    // 144x512
  unsigned short* h_bf   = (unsigned short*)alloc((size_t)ROWS * DMODEL * 2);
  float*          xz     = (float*)alloc((size_t)ROWS * DXZ * 4);
  float*          xc     = (float*)alloc((size_t)ROWS * DINNER * 4);
  unsigned short* xc_bf  = (unsigned short*)alloc((size_t)ROWS * DINNER * 2);
  float*          xdbl   = (float*)alloc((size_t)ROWS * NDBL * 4);
  unsigned short* dtin   = (unsigned short*)alloc((size_t)ROWS * DTRANK * 2);
  float*          dtp    = (float*)alloc((size_t)ROWS * DINNER * 4);             // raw -> softplus in-place
  unsigned short* y_bf   = (unsigned short*)alloc((size_t)ROWS * DINNER * 2);
  float*          yo     = (float*)alloc((size_t)ROWS * DMODEL * 4);
  unsigned short* yo_bf  = (unsigned short*)alloc((size_t)ROWS * DMODEL * 2);

  // Weight transpose + bf16 convert
  transpose_bf16_kernel<<<CEILDIV(DMODEL * DXZ, 256), 256, 0, stream>>>(in_proj, Wt_in, DMODEL, DXZ);
  transpose_bf16_kernel<<<CEILDIV(DINNER * NDBL, 256), 256, 0, stream>>>(x_proj, Wt_xp, DINNER, NDBL);
  transpose_bf16_kernel<<<CEILDIV(DTRANK * DINNER, 256), 256, 0, stream>>>(dt_proj, Wt_dt, DTRANK, DINNER);
  transpose_bf16_kernel<<<CEILDIV(DINNER * DMODEL, 256), 256, 0, stream>>>(out_proj, Wt_op, DINNER, DMODEL);
  transpose_bf16_kernel<<<CEILDIV(DMODEL * NOUT, 256), 256, 0, stream>>>(w1, Wt_w1, DMODEL, NOUT);

  // h = tanh(x@w0+b0)
  embed_kernel<<<CEILDIV(ROWS * DMODEL, 256), 256, 0, stream>>>(x, w0, b0, h_bf);

  // xz = h @ in_proj  (M=8192,N=2048,K=512): 512 Mtiles x 32 groups of 4 Ntiles
  wmma_gemm_bf16_k<4><<<(512 * 32) / 8, 256, 0, stream>>>(
      h_bf, Wt_in, xz, 512, 32, DMODEL, DMODEL, DMODEL, DXZ);

  // depthwise conv + SiLU
  conv_silu_kernel<<<CEILDIV(ROWS * DINNER, 256), 256, 0, stream>>>(xz, conv_w, conv_b, xc, xc_bf);

  // x_dbl = xc @ x_proj  (M=8192,N=64,K=1024)
  wmma_gemm_bf16_k<4><<<(512 * 1) / 8, 256, 0, stream>>>(
      xc_bf, Wt_xp, xdbl, 512, 1, DINNER, DINNER, DINNER, NDBL);

  slice_dt_bf16_kernel<<<CEILDIV(ROWS * DTRANK, 256), 256, 0, stream>>>(xdbl, dtin);

  // dtp = dt_in @ dt_proj  (M=8192,N=1024,K=32)
  wmma_gemm_bf16_k<4><<<(512 * 16) / 8, 256, 0, stream>>>(
      dtin, Wt_dt, dtp, 512, 16, DTRANK, DTRANK, DTRANK, DINNER);

  softplus_kernel<<<CEILDIV(ROWS * DINNER, 256), 256, 0, stream>>>(dtp, dt_pb);

  // selective scan + gate -> y (bf16)
  scan_kernel<<<(B_SEQ * DINNER * DSTATE) / 256, 256, 0, stream>>>(
      dtp, xc, xdbl, xz, A_log, Dv, y_bf);

  // yo = y @ out_proj  (M=8192,N=512,K=1024)
  wmma_gemm_bf16_k<4><<<(512 * 8) / 8, 256, 0, stream>>>(
      y_bf, Wt_op, yo, 512, 8, DINNER, DINNER, DINNER, DMODEL);

  cvt_bf16_kernel<<<CEILDIV(ROWS * DMODEL, 256), 256, 0, stream>>>(yo, yo_bf, ROWS * DMODEL);

  // logits = yo @ w1  (M=8192,N=144,K=512) -> straight into d_out
  wmma_gemm_bf16_k<1><<<(512 * 9) / 8, 256, 0, stream>>>(
      yo_bf, Wt_w1, out, 512, 9, DMODEL, DMODEL, DMODEL, NOUT);

  // softmax(+b1) in place, one wave per row
  softmax_kernel<<<ROWS / 8, 256, 0, stream>>>(out, b1);
}
